// MultiheadSampler_11562051960832
// MI455X (gfx1250) — matrix operand order repaired and checked by reading, not compiled, not run on hardware
//
#include <hip/hip_runtime.h>
#include <stdint.h>

#define BB 64
#define HH 4
#define VV 128000
#define NN 256
#define WW 200
#define TPB 1024
#define CHUNK 2048            // floats per staged chunk (8KB)
#define NFULL 62              // 62*2048 = 126976
#define TAILN 1024            // 128000 - 126976
#define CAND 2048             // max top-k candidates kept (k <= 1024 + ties)

typedef __attribute__((ext_vector_type(16))) _Float16 v16h;
typedef __attribute__((ext_vector_type(8)))  float    v8f;

// Monotonic float -> u32 key (ascending order preserved)
__device__ __forceinline__ uint32_t fkey(float x) {
  uint32_t b = __float_as_uint(x);
  return (b & 0x80000000u) ? ~b : (b | 0x80000000u);
}

// ---- CDNA5 async global->LDS copy (16B per lane), tracked by ASYNCcnt ----
__device__ __forceinline__ void async_cp16(uint32_t lds_addr, const float* g) {
  asm volatile("global_load_async_to_lds_b128 %0, %1, off"
               :: "v"(lds_addr), "v"((uint64_t)(uintptr_t)g)
               : "memory");
}
__device__ __forceinline__ void wait_async(int pending) {
  if (pending) asm volatile("s_wait_asynccnt 0x1" ::: "memory");
  else         asm volatile("s_wait_asynccnt 0x0" ::: "memory");
}

// Stream one row of V floats through double-buffered async LDS staging.
// fn(value, global_index) called for every element. Ends with a barrier.
template <typename F>
__device__ void stream_row(const float* __restrict__ row, float* stage, F&& fn) {
  const int tid = threadIdx.x;
  float* buf0 = stage;
  float* buf1 = stage + CHUNK;
  if (tid < CHUNK / 4)
    async_cp16((uint32_t)(uintptr_t)(buf0 + 4 * tid), row + 4 * tid);
  for (int c = 0; c < NFULL; ++c) {
    const int nx = c + 1;
    float* cur = (c & 1) ? buf1 : buf0;
    if (nx < NFULL && tid < CHUNK / 4)
      async_cp16((uint32_t)(uintptr_t)(((nx & 1) ? buf1 : buf0) + 4 * tid),
                 row + (size_t)nx * CHUNK + 4 * tid);
    wait_async(nx < NFULL ? 1 : 0);
    __syncthreads();                       // chunk c visible to all waves
    const float2 v2 = *reinterpret_cast<const float2*>(cur + 2 * tid);
    const int base = c * CHUNK + 2 * tid;
    fn(v2.x, base);
    fn(v2.y, base + 1);
    __syncthreads();                       // done reading before buffer reuse
  }
  { // tail: 1024 elements, one per thread
    const int idx = NFULL * CHUNK + tid;
    fn(row[idx], idx);
  }
  __syncthreads();
}

__global__ void __launch_bounds__(TPB)
sampler_kernel(const float* __restrict__ logits,
               const int*   __restrict__ tokens,
               const float* __restrict__ temps,
               const float* __restrict__ topps,
               const int*   __restrict__ topks,
               const float* __restrict__ reps,
               float* __restrict__ out_ids,
               float* __restrict__ out_probs) {
  __shared__ __align__(16) float stage[2 * CHUNK];   // 16 KB async staging
  __shared__ uint32_t mask[VV / 32];                 // 16 KB seen-token bitmask
  __shared__ uint32_t hist[256];
  __shared__ float cval[CAND];
  __shared__ int   cidx[CAND];
  __shared__ float escan[CAND + CAND / 64];          // padded (stride-65 segs)
  __shared__ float totals[32];
  __shared__ float Gs[32];
  __shared__ uint32_t sh_prefix;
  __shared__ int sh_kk, sh_cnt, sh_R, sh_sel;
  __shared__ float sh_target;

  const int row = blockIdx.x;
  const int tid = threadIdx.x;
  const float* rowp = logits + (size_t)row * VV;
  const float temp = temps[row];
  const float rp   = reps[row];
  const float pp   = topps[row];
  const int   k0   = topks[row];

  for (int i = tid; i < VV / 32; i += TPB) mask[i] = 0u;
  if (tid < 256) hist[tid] = 0u;
  if (tid == 0) { sh_prefix = 0u; sh_kk = k0; sh_cnt = 0; }
  __syncthreads();
  for (int i = tid; i < WW; i += TPB) {
    int t = tokens[row * WW + i];
    if ((unsigned)t < (unsigned)VV) atomicOr(&mask[t >> 5], 1u << (t & 31));
  }
  __syncthreads();

  // repetition penalty + temperature (recomputed each pass; L2-resident input)
  auto scaled_of = [&](float v, int idx) -> float {
    bool seen = (mask[idx >> 5] >> (idx & 31)) & 1u;
    float r = seen ? rp : 1.0f;
    float s = (v > 0.0f) ? (v / r) : (v * r);
    return s / temp;
  };

  // ---- 4-pass radix select: exact key of the k-th largest scaled logit ----
  for (int p = 0; p < 4; ++p) {
    if (p > 0) { if (tid < 256) hist[tid] = 0u; __syncthreads(); }
    const int matchShift = 32 - 8 * p;
    const int binShift   = 24 - 8 * p;
    const uint32_t pref  = sh_prefix;
    stream_row(rowp, stage, [&](float v, int idx) {
      uint32_t key = fkey(scaled_of(v, idx));
      bool ok = (p == 0) || ((key >> matchShift) == (pref >> matchShift));
      if (ok) atomicAdd(&hist[(key >> binShift) & 255u], 1u);
    });
    if (tid == 0) {
      uint32_t cum = 0; int kk = sh_kk;
      for (int b = 255; b >= 0; --b) {
        uint32_t h = hist[b];
        if (cum + h >= (uint32_t)kk) {
          sh_prefix = pref | ((uint32_t)b << binShift);
          sh_kk = kk - (int)cum;
          break;
        }
        cum += h;
      }
    }
    __syncthreads();
  }
  const uint32_t kthkey = sh_prefix;

  // ---- collect top-k candidates (value >= kth) ----
  stream_row(rowp, stage, [&](float v, int idx) {
    float s = scaled_of(v, idx);
    if (fkey(s) >= kthkey) {
      int pos = atomicAdd(&sh_cnt, 1);
      if (pos < CAND) { cval[pos] = s; cidx[pos] = idx; }
    }
  });
  const int M = min(sh_cnt, CAND);
  for (int j = M + tid; j < CAND; j += TPB) { cval[j] = -3.4e38f; cidx[j] = 0; }
  __syncthreads();

  // ---- bitonic sort descending (2048 elems, 1024 threads) ----
  for (unsigned k = 2; k <= CAND; k <<= 1) {
    for (unsigned j = k >> 1; j > 0; j >>= 1) {
      for (int s = 0; s < 2; ++s) {
        unsigned i = tid + s * TPB;
        unsigned ixj = i ^ j;
        if (ixj > i) {
          bool desc = ((i & k) == 0);
          float a = cval[i], b2 = cval[ixj];
          if (desc ? (a < b2) : (a > b2)) {
            cval[i] = b2; cval[ixj] = a;
            int ti = cidx[i]; cidx[i] = cidx[ixj]; cidx[ixj] = ti;
          }
        }
      }
      __syncthreads();
    }
  }

  // ---- exp + hierarchical inclusive cumsum (segments + WMMA totals scan) ----
  const float m0 = cval[0];
  for (int s = 0; s < 2; ++s) {
    int j = tid + s * TPB;
    float e = (j < M) ? __expf(cval[j] - m0) : 0.0f;
    escan[j + (j >> 6)] = e;                      // stride-65 padded segments
  }
  __syncthreads();
  if (tid < 32) {                                  // 32 serial segment scans
    float run = 0.0f;
    const int pb = tid * 65;
    for (int i2 = 0; i2 < 64; ++i2) { run += escan[pb + i2]; escan[pb + i2] = run; }
    totals[tid] = run;
  }
  __syncthreads();

#if __has_builtin(__builtin_amdgcn_wmma_f32_16x16x32_f16)
  // Scan 32 segment totals with one v_wmma_f32_16x16x32_f16:
  // A = [Tri16 | 0] (16x32), B cols 0/1 = t_lo/t_hi -> C(:,0/1) = prefixes.
  if (tid < 32) {
    const int lane = tid;
    const int mr = lane & 15;
    const bool hi = lane >= 16;
    v16h a, b;
#pragma unroll
    for (int h = 0; h < 16; ++h) {
      int K = h + (hi ? 16 : 0);                           // A: contiguous k = 16*half + h
      a[h] = (_Float16)((K < 16 && K <= mr) ? 1.0f : 0.0f);
      int KB = h + (hi ? 16 : 0);                          // B: same contiguous k order as A
      float tv = (KB < 16 && mr < 2) ? totals[16 * mr + KB] : 0.0f;
      b[h] = (_Float16)tv;
    }
    v8f cacc = {};
    cacc = __builtin_amdgcn_wmma_f32_16x16x32_f16(
        false, a, false, b, (short)0, cacc, false, false);
    if (mr < 2) {                                          // cols 0,1 -> Gs
      int base = mr * 16 + (hi ? 8 : 0);
#pragma unroll
      for (int r = 0; r < 8; ++r) Gs[base + r] = cacc[r];
    }
  }
#else
  if (tid == 0) { float run = 0.0f; for (int i2 = 0; i2 < 32; ++i2) { run += totals[i2]; Gs[i2] = run; } }
#endif
  __syncthreads();
  if (tid == 0) { float slo = Gs[15]; for (int i2 = 16; i2 < 32; ++i2) Gs[i2] += slo; }
  __syncthreads();
  for (int s = 0; s < 2; ++s) {                   // add segment offsets
    int j = tid + s * TPB;
    int seg = j >> 6;
    if (seg > 0) escan[j + seg] += Gs[seg - 1];
  }
  __syncthreads();

#define CPS(j) escan[(j) + ((j) >> 6)]
  // ---- top-p: keep rank j iff prefix-prob before j < top_p ----
  const float Etot = CPS(M - 1);
  const float limit = pp * Etot;
  if (tid == 0) { sh_R = 1; }
  __syncthreads();
  for (int s = 0; s < 2; ++s) {
    int j = tid + s * TPB;
    if (j >= 1 && j < M && CPS(j - 1) < limit) atomicAdd(&sh_R, 1);
  }
  __syncthreads();
  const int R = min(sh_R, M);
  const float Z = CPS(R - 1);
  const float thresh = cval[R - 1];
  const float invZ = 1.0f / Z;

  // ---- deterministic inverse-CDF sample ----
  if (tid == 0) {
    uint32_t h = 0x9E3779B9u * (uint32_t)(row + 1) ^ 42u;
    h ^= h >> 16; h *= 0x85EBCA6Bu; h ^= h >> 13; h *= 0xC2B2AE35u; h ^= h >> 16;
    float u = ((h >> 8) + 0.5f) * (1.0f / 16777216.0f);
    sh_target = u * Z;
    sh_sel = R - 1;
  }
  __syncthreads();
  const float target = sh_target;
  for (int s = 0; s < 2; ++s) {
    int j = tid + s * TPB;
    if (j < R) {
      float cj = CPS(j);
      float cp = (j == 0) ? 0.0f : CPS(j - 1);
      if (cj >= target && cp < target) atomicMin(&sh_sel, j);
    }
  }
  __syncthreads();
  if (tid == 0) out_ids[row] = (float)cidx[sh_sel];

  // ---- final streaming pass: write probs (NT stores, single threshold) ----
  float* oprow = out_probs + (size_t)row * VV;
  stream_row(rowp, stage, [&](float v, int idx) {
    float s = scaled_of(v, idx);
    float pr = (s >= thresh) ? __expf(s - m0) * invZ : 0.0f;
    __builtin_nontemporal_store(pr, &oprow[idx]);
  });
#undef CPS
}

extern "C" void kernel_launch(void* const* d_in, const int* in_sizes, int n_in,
                              void* d_out, int out_size, void* d_ws, size_t ws_size,
                              hipStream_t stream) {
  const float* logits = (const float*)d_in[0];
  const int*   tokens = (const int*)d_in[1];
  const float* temps  = (const float*)d_in[2];
  const float* topps  = (const float*)d_in[3];
  const int*   topks  = (const int*)d_in[4];
  const float* reps   = (const float*)d_in[5];
  float* out_ids   = (float*)d_out;        // 256 ids, then 256*128000 probs
  float* out_probs = out_ids + NN;
  (void)in_sizes; (void)n_in; (void)out_size; (void)d_ws; (void)ws_size;
  sampler_kernel<<<dim3(NN), dim3(TPB), 0, stream>>>(
      logits, tokens, temps, topps, topks, reps, out_ids, out_probs);
}
